// VQVAE_84877143703650
// MI455X (gfx1250) — matrix operand order repaired and checked by reading, not compiled, not run on hardware
//
#include <hip/hip_runtime.h>
#include <hip/hip_bf16.h>

// ---------------------------------------------------------------------------
// VQ-VAE forward for MI455X (gfx1250, wave32).
// Conv layers + VQ distance run on v_wmma_f32_16x16x32_f16 (f16 in, f32 acc).
// Weights are pre-packed to f16 tap-major and staged into LDS with
// global_load_async_to_lds_b128 (ASYNCcnt). Codebook staged in LDS for VQ.
// ---------------------------------------------------------------------------

typedef __attribute__((ext_vector_type(16))) _Float16 v16h;
typedef __attribute__((ext_vector_type(8)))  float    v8f;

union AFrag { v16h v; _Float16 h[16]; unsigned int u[8]; };
union CFrag { v8f  v; float    f[8];  };

#define WMMA_F16(A, B, C) \
    __builtin_amdgcn_wmma_f32_16x16x32_f16(false, (A), false, (B), (short)0, (C), false, false)

// ---------------------------------------------------------------------------
// Generic implicit-GEMM Conv1d, reduction packed TAP-MAJOR (f16 weights):
//   Wp[co][kt*Cin + ci],  Y[b,co,l*oy+ooff] = bias + sum_kt sum_ci Wp * X[b,ci,l*SX+d(kt)]
// KT/SX/RELU template constants; Cin multiple of 32. RELU applies max(x,0) to
// the conv input (fused residual-branch ReLU).
// Block = 256 threads = 8 waves; wave computes 16(co) x 16(l); block covers
// 16 co x 128 l. Grid: (Lout/128, Cout/16, B).
// Weight tile (16 x R f16) async-copied Global->LDS once per block.
// ---------------------------------------------------------------------------
template <int KT, int SX, int RELU>
__global__ void __launch_bounds__(256)
conv1d_wmma(const float* __restrict__ X, const _Float16* __restrict__ Wp,
            const float* __restrict__ bias, float* __restrict__ Y,
            int Cin, int Lin, int Cout, int LoutPhys,
            int d0, int d1, int d2, int d3, int oy, int ooff)
{
    const int lane = threadIdx.x & 31;
    const int wid  = threadIdx.x >> 5;
    const int b    = blockIdx.z;
    const int co0  = blockIdx.y * 16;
    const int l0   = blockIdx.x * 128 + wid * 16;

    const int dk[4] = {d0, d1, d2, d3};
    const int R = KT * Cin;

    // ---- async-copy the contiguous 16 x R f16 weight slab into LDS --------
    __shared__ __align__(16) _Float16 Wlds[16 * 768];     // max R = 768 -> 24 KB
    {
        const char* __restrict__ gsrc = (const char*)(Wp + (size_t)co0 * R);
        unsigned lbase = (unsigned)(uintptr_t)(&Wlds[0]);
        int units = 2 * R;                                // 16-byte units in slab
        for (int u = threadIdx.x; u < units; u += 256) {
            unsigned           laddr = lbase + (unsigned)(u * 16);
            unsigned long long gaddr = (unsigned long long)(uintptr_t)(gsrc + (size_t)u * 16);
            asm volatile("global_load_async_to_lds_b128 %0, %1, off"
                         :: "v"(laddr), "v"(gaddr) : "memory");
        }
        asm volatile("s_wait_asynccnt 0" ::: "memory");
    }
    __syncthreads();

    const float* __restrict__ Xb = X + (size_t)b * Cin * Lin;

    const int khalfA = (lane >> 4) * 8;     // A layout K-half select
    const int khB    = (lane >> 4) * 16;    // B layout K-half select
    const int ncol   = l0 + (lane & 15);
    const _Float16* __restrict__ wl = &Wlds[(lane & 15) * R];

    CFrag acc;
#pragma unroll
    for (int i = 0; i < 8; ++i) acc.f[i] = 0.0f;

#pragma unroll
    for (int kt = 0; kt < KT; ++kt) {
        int  p     = ncol * SX + dk[kt];
        bool valid = (p >= 0) && (p < Lin);
        int  pc    = valid ? p : 0;
        float msk  = valid ? 1.0f : 0.0f;
        const float* __restrict__ xcol = Xb + pc;
        const int wbase = kt * Cin;

        for (int c0 = 0; c0 < Cin; c0 += 32) {
            if (c0 + 32 < Cin)
                __builtin_prefetch(xcol + (size_t)(c0 + 32 + khB) * Lin, 0, 1);

            AFrag a, bb;
#pragma unroll
            for (int rr = 0; rr < 8; ++rr) {
                // A, contiguous k order (matches B): lanes0-15 K0..15, lanes16-31 K16..31 (pairs per VGPR).
                int k = khB + rr * 2;
                a.u[rr] = *(const unsigned int*)&wl[wbase + c0 + k];   // ds_load
            }
#pragma unroll
            for (int rr = 0; rr < 8; ++rr) {
                // B, contiguous k order (matches A): lanes0-15 K0..15, lanes16-31 K16..31.
                int ci = c0 + khB + rr * 2;
                float x0 = xcol[(size_t)ci * Lin];
                float x1 = xcol[(size_t)(ci + 1) * Lin];
                if (RELU) { x0 = fmaxf(x0, 0.0f); x1 = fmaxf(x1, 0.0f); }
                bb.h[2 * rr]     = (_Float16)(msk * x0);
                bb.h[2 * rr + 1] = (_Float16)(msk * x1);
            }
            acc.v = WMMA_F16(a.v, bb.v, acc.v);
        }
    }

    // C/D layout: reg rr, lane: M = rr + 8*(lane>=16), N = lane&15
    const int mo = (lane >> 4) * 8;
    float* __restrict__ Yb = Y + (size_t)b * Cout * LoutPhys;
#pragma unroll
    for (int rr = 0; rr < 8; ++rr) {
        int m = co0 + mo + rr;
        float v = acc.f[rr] + (bias ? bias[m] : 0.0f);
        Yb[(size_t)m * LoutPhys + (size_t)ncol * oy + ooff] = v;
    }
}

// ---------------------------------------------------------------------------
// Tap-major f16 weight repack: w[Cout][Cin][K] -> wp[Cout][K*Cin] (wp[co][k*Cin+ci])
// ---------------------------------------------------------------------------
__global__ void __launch_bounds__(256)
pack_conv(const float* __restrict__ w, _Float16* __restrict__ wp,
          int Cin, int Cout, int K)
{
    int i = blockIdx.x * 256 + threadIdx.x;
    if (i >= Cout * Cin * K) return;
    int co  = i / (Cin * K);
    int rem = i - co * (Cin * K);
    int ci  = rem / K;
    int k   = rem - ci * K;
    wp[(size_t)co * (Cin * K) + k * Cin + ci] = (_Float16)w[i];
}

// ---------------------------------------------------------------------------
// ConvTranspose1d(k=4,s=2,p=1) repack, tap-major f16: w torch [Cin, Cout, 4]
//   -> wp[phase][Cout][kt*Cin + ci]
//   phase0 (even o=2t): kt0 = (k=1,d=0),  kt1 = (k=3,d=-1)
//   phase1 (odd  o)   : kt0 = (k=0,d=+1), kt1 = (k=2,d=0)
// ---------------------------------------------------------------------------
__global__ void __launch_bounds__(256)
packT_weights(const float* __restrict__ w, _Float16* __restrict__ wp,
              int Cin, int Cout)
{
    int i = blockIdx.x * 256 + threadIdx.x;
    if (i >= Cin * Cout) return;
    int ci = i / Cout, co = i - ci * Cout;
    const float* __restrict__ ws = w + ((size_t)ci * Cout + co) * 4;
    size_t R     = (size_t)Cin * 2;
    size_t row   = (size_t)co * R;
    size_t phase = (size_t)Cout * R;
    wp[row + ci]               = (_Float16)ws[1];
    wp[row + Cin + ci]         = (_Float16)ws[3];
    wp[phase + row + ci]       = (_Float16)ws[0];
    wp[phase + row + Cin + ci] = (_Float16)ws[2];
}

// ---------------------------------------------------------------------------
// Encoder conv1: [32,1,8192] -> [32,128,4096], k=4 s=2 p=1 (Cin=1: direct VALU)
// ---------------------------------------------------------------------------
__global__ void __launch_bounds__(256)
conv1_direct(const float* __restrict__ x, const float* __restrict__ w,
             const float* __restrict__ b1, float* __restrict__ y)
{
    int i  = blockIdx.x * 256 + threadIdx.x;        // 32*128*4096
    int l  = i & 4095;
    int co = (i >> 12) & 127;
    int bb = i >> 19;
    const float* __restrict__ xb = x + (size_t)bb * 8192;
    float acc = b1[co];
    int p0 = 2 * l - 1;
#pragma unroll
    for (int k = 0; k < 4; ++k) {
        int p = p0 + k;
        if (p >= 0 && p < 8192) acc += w[co * 4 + k] * xb[p];
    }
    y[i] = acc;
}

// ---------------------------------------------------------------------------
// BatchNorm (training mode): stats over (B, L) per channel, biased variance.
// Lc = 1<<lsh, C power of two.
// ---------------------------------------------------------------------------
__global__ void __launch_bounds__(256)
bn_stats(const float* __restrict__ X, float* __restrict__ stats,
         int C, int lsh, int Bn)
{
    int c  = blockIdx.x;
    int Lc = 1 << lsh;
    int N  = Bn << lsh;
    float s1 = 0.0f, s2 = 0.0f;
    for (int i = threadIdx.x; i < N; i += 256) {
        int b = i >> lsh, l = i & (Lc - 1);
        float v = X[((size_t)b * C + c) * Lc + l];
        s1 += v; s2 += v * v;
    }
    __shared__ float r1[256], r2[256];
    r1[threadIdx.x] = s1; r2[threadIdx.x] = s2;
    __syncthreads();
    for (int s = 128; s > 0; s >>= 1) {
        if (threadIdx.x < s) {
            r1[threadIdx.x] += r1[threadIdx.x + s];
            r2[threadIdx.x] += r2[threadIdx.x + s];
        }
        __syncthreads();
    }
    if (threadIdx.x == 0) {
        float m   = r1[0] / (float)N;
        float var = r2[0] / (float)N - m * m;
        stats[2 * c]     = m;
        stats[2 * c + 1] = rsqrtf(var + 1e-5f);
    }
}

__global__ void __launch_bounds__(256)
bn_apply(const float* __restrict__ X, const float* __restrict__ RES,
         float* __restrict__ Y, const float* __restrict__ g,
         const float* __restrict__ bt, const float* __restrict__ stats,
         int cmask, int lsh, int relu, int total)
{
    int i = blockIdx.x * 256 + threadIdx.x;
    if (i >= total) return;
    int c = (i >> lsh) & cmask;
    float v = (X[i] - stats[2 * c]) * stats[2 * c + 1] * g[c] + bt[c];
    if (relu) v = fmaxf(v, 0.0f);
    if (RES)  v += RES[i];
    Y[i] = v;
}

// ---------------------------------------------------------------------------
// Vector quantizer. Faithful torch .view(-1, 64): rows are 64 consecutive
// floats of the contiguous [B, D, L'] buffer. 65536 rows x 512 codes x K=64.
// The f16 codebook (64 KB) is staged in LDS once per block, shared by 8 waves.
// ---------------------------------------------------------------------------
__global__ void __launch_bounds__(256)
enorm_k(const float* __restrict__ E, float* __restrict__ en)
{
    int j = blockIdx.x * 256 + threadIdx.x;
    if (j < 512) {
        const float* __restrict__ e = E + (size_t)j * 64;
        float s = 0.0f;
#pragma unroll
        for (int d = 0; d < 64; ++d) s += e[d] * e[d];
        en[j] = s;
    }
}

__global__ void __launch_bounds__(256)
vq_argmin(const float* __restrict__ Z, const float* __restrict__ E,
          const float* __restrict__ enorm, int* __restrict__ idx_out,
          float* __restrict__ idxf_out)
{
    __shared__ _Float16 Elds[512 * 64];              // 64 KB
    for (int i = threadIdx.x; i < 512 * 64; i += 256)
        Elds[i] = (_Float16)E[i];
    __syncthreads();

    const int lane    = threadIdx.x & 31;
    const int wid     = (blockIdx.x * 256 + threadIdx.x) >> 5;   // global wave id
    const int rowbase = wid * 16;

    // A fragments (Z rows), K=64 -> two 32-chunks
    const int m      = rowbase + (lane & 15);
    const int khalfA = (lane >> 4) * 8;
    const float* __restrict__ zr = Z + (size_t)m * 64;
    AFrag a0, a1;
#pragma unroll
    for (int rr = 0; rr < 8; ++rr) {
        int k = (lane >> 4) * 16 + rr * 2;
        a0.h[2 * rr]     = (_Float16)zr[k];
        a0.h[2 * rr + 1] = (_Float16)zr[k + 1];
        a1.h[2 * rr]     = (_Float16)zr[32 + k];
        a1.h[2 * rr + 1] = (_Float16)zr[32 + k + 1];
    }

    const int nl  = lane & 15;
    const int khB = (lane >> 4) * 16;

    float best[8]; int bidx[8];
#pragma unroll
    for (int rr = 0; rr < 8; ++rr) { best[rr] = 3.4e38f; bidx[rr] = 0; }

    for (int jt = 0; jt < 32; ++jt) {
        int j = jt * 16 + nl;
        const _Float16* __restrict__ er = &Elds[j * 64];   // B[k][n] = E[j=n][k]
        AFrag b0, b1;
#pragma unroll
        for (int rr = 0; rr < 8; ++rr) {
            int k = khB + rr * 2;
            b0.u[rr] = *(const unsigned int*)&er[k];        // ds_load_b128 merged
            b1.u[rr] = *(const unsigned int*)&er[32 + k];
        }
        CFrag acc;
#pragma unroll
        for (int rr = 0; rr < 8; ++rr) acc.f[rr] = 0.0f;
        acc.v = WMMA_F16(a0.v, b0.v, acc.v);
        acc.v = WMMA_F16(a1.v, b1.v, acc.v);

        float en = enorm[j];
#pragma unroll
        for (int rr = 0; rr < 8; ++rr) {
            float dd = en - 2.0f * acc.f[rr];   // ||x||^2 constant per row, omitted
            if (dd < best[rr]) { best[rr] = dd; bidx[rr] = j; }
        }
    }

    // argmin across the 16 lanes holding the same row (xor<16 stays in half)
#pragma unroll
    for (int off = 8; off >= 1; off >>= 1) {
#pragma unroll
        for (int rr = 0; rr < 8; ++rr) {
            float ob = __shfl_xor(best[rr], off, 32);
            int   oi = __shfl_xor(bidx[rr], off, 32);
            if (ob < best[rr] || (ob == best[rr] && oi < bidx[rr])) {
                best[rr] = ob; bidx[rr] = oi;
            }
        }
    }

    if ((lane & 15) == 0) {
        int rbase = rowbase + ((lane >> 4) * 8);
#pragma unroll
        for (int rr = 0; rr < 8; ++rr) {
            idx_out[rbase + rr]  = bidx[rr];
            idxf_out[rbase + rr] = (float)bidx[rr];
        }
    }
}

__global__ void __launch_bounds__(256)
vq_gather(const float* __restrict__ Z, const float* __restrict__ E,
          const int* __restrict__ idx, float* __restrict__ zq,
          float* __restrict__ loss)
{
    int i   = blockIdx.x * 256 + threadIdx.x;   // 65536*64
    int row = i >> 6, d = i & 63;
    int j   = idx[row];
    float q    = E[(size_t)j * 64 + d];
    float diff = q - Z[i];
    zq[i] = q;                                   // straight-through value

    __shared__ float r[256];
    r[threadIdx.x] = diff * diff;
    __syncthreads();
    for (int s = 128; s > 0; s >>= 1) {
        if (threadIdx.x < s) r[threadIdx.x] += r[threadIdx.x + s];
        __syncthreads();
    }
    if (threadIdx.x == 0) atomicAdd(loss, r[0]);
}

__global__ void finalize_loss(const float* __restrict__ loss, float* __restrict__ out)
{
    // vq_loss = q_loss + 0.25*e_loss = 1.25 * mean((q - z_e)^2)
    *out = 1.25f * (*loss) / 4194304.0f;
}

// ---------------------------------------------------------------------------
// Decoder convT2: [32,128,4096] -> [32,1,8192] (Cout=1: direct VALU)
// w: torch [128, 1, 4]
// ---------------------------------------------------------------------------
__global__ void __launch_bounds__(256)
convT2_direct(const float* __restrict__ X, const float* __restrict__ w,
              const float* __restrict__ bt, float* __restrict__ out)
{
    int i = blockIdx.x * 256 + threadIdx.x;     // 32*8192
    int o = i & 8191;
    int b = i >> 13;
    int t = o >> 1;
    const float* __restrict__ Xb = X + (size_t)b * 128 * 4096;
    float acc = bt[0];
    if ((o & 1) == 0) {
        for (int ci = 0; ci < 128; ++ci) {
            const float* __restrict__ xr = Xb + (size_t)ci * 4096;
            acc += xr[t] * w[ci * 4 + 1];
            if (t > 0) acc += xr[t - 1] * w[ci * 4 + 3];
        }
    } else {
        for (int ci = 0; ci < 128; ++ci) {
            const float* __restrict__ xr = Xb + (size_t)ci * 4096;
            if (t + 1 < 4096) acc += xr[t + 1] * w[ci * 4 + 0];
            acc += xr[t] * w[ci * 4 + 2];
        }
    }
    out[i] = acc;
}

// ---------------------------------------------------------------------------
// Host orchestration
// ---------------------------------------------------------------------------
extern "C" void kernel_launch(void* const* d_in, const int* in_sizes, int n_in,
                              void* d_out, int out_size, void* d_ws, size_t ws_size,
                              hipStream_t stream)
{
    (void)in_sizes; (void)n_in; (void)out_size; (void)ws_size;

    // ---- inputs, in setup_inputs() insertion order (depth-first) ----
    const float* x       = (const float*)d_in[0];
    const float* e_w1    = (const float*)d_in[1];
    const float* e_b1    = (const float*)d_in[2];
    const float* e_bn1g  = (const float*)d_in[3];
    const float* e_bn1b  = (const float*)d_in[4];
    const float* e_w2    = (const float*)d_in[5];
    const float* e_b2    = (const float*)d_in[6];
    const float* e_bn2g  = (const float*)d_in[7];
    const float* e_bn2b  = (const float*)d_in[8];
    const float* e_w3    = (const float*)d_in[9];
    const float* e_b3    = (const float*)d_in[10];
    const float* e_bn3g  = (const float*)d_in[11];
    const float* e_bn3b  = (const float*)d_in[12];
    const float* er1_w1  = (const float*)d_in[13];
    const float* er1_g1  = (const float*)d_in[14];
    const float* er1_b1  = (const float*)d_in[15];
    const float* er1_w2  = (const float*)d_in[16];
    const float* er1_g2  = (const float*)d_in[17];
    const float* er1_b2  = (const float*)d_in[18];
    const float* er2_w1  = (const float*)d_in[19];
    const float* er2_g1  = (const float*)d_in[20];
    const float* er2_b1  = (const float*)d_in[21];
    const float* er2_w2  = (const float*)d_in[22];
    const float* er2_g2  = (const float*)d_in[23];
    const float* er2_b2  = (const float*)d_in[24];
    const float* e_w4    = (const float*)d_in[25];
    const float* e_b4    = (const float*)d_in[26];
    const float* Ecb     = (const float*)d_in[27];
    const float* d_w1    = (const float*)d_in[28];
    const float* d_b1    = (const float*)d_in[29];
    const float* d_bn1g  = (const float*)d_in[30];
    const float* d_bn1b  = (const float*)d_in[31];
    const float* dr1_w1  = (const float*)d_in[32];
    const float* dr1_g1  = (const float*)d_in[33];
    const float* dr1_b1  = (const float*)d_in[34];
    const float* dr1_w2  = (const float*)d_in[35];
    const float* dr1_g2  = (const float*)d_in[36];
    const float* dr1_b2  = (const float*)d_in[37];
    const float* dr2_w1  = (const float*)d_in[38];
    const float* dr2_g1  = (const float*)d_in[39];
    const float* dr2_b1  = (const float*)d_in[40];
    const float* dr2_w2  = (const float*)d_in[41];
    const float* dr2_g2  = (const float*)d_in[42];
    const float* dr2_b2  = (const float*)d_in[43];
    const float* d_wt1   = (const float*)d_in[44];
    const float* d_bt1   = (const float*)d_in[45];
    const float* d_bn2g  = (const float*)d_in[46];
    const float* d_bn2b  = (const float*)d_in[47];
    const float* d_wt2   = (const float*)d_in[48];
    const float* d_bt2   = (const float*)d_in[49];

    float* out = (float*)d_out;   // [0,262144) x_recon, [262144] loss, [262145,..) idx

    // ---- workspace arena ----
    char*  ws  = (char*)d_ws;
    size_t off = 0;
    auto alloc = [&](size_t nbytes) -> void* {
        void* p = ws + off;
        off += (nbytes + 255) & ~(size_t)255;
        return p;
    };
    const size_t N256 = (size_t)32 * 256 * 2048;   // 16,777,216
    float* P128 = (float*)alloc(N256 * 4);         // [32,128,4096] (same count)
    float* A    = (float*)alloc(N256 * 4);
    float* Bf   = (float*)alloc(N256 * 4);
    float* Cf   = (float*)alloc(N256 * 4);
    float* Z1   = (float*)alloc((size_t)32 * 64 * 2048 * 4);
    float* Z2   = (float*)alloc((size_t)32 * 64 * 2048 * 4);
    int*   idxI = (int*)alloc(65536 * 4);
    float* stats  = (float*)alloc(512 * 4);
    float* enormB = (float*)alloc(512 * 4);
    float* lossB  = (float*)alloc(256);
    _Float16* wpk  = (_Float16*)alloc(196608 * 2);           // f16 packed-weight scratch
    _Float16* wtp1 = (_Float16*)alloc((size_t)2 * 128 * 512 * 2);

    const int TOT256 = 16777216;   // elements of [32,256,2048] / [32,128,4096]
    const int GB256  = TOT256 / 256;

    auto pack = [&](const float* w, int Cin, int Cout, int K) {
        int n = Cout * Cin * K;
        pack_conv<<<(n + 255) / 256, 256, 0, stream>>>(w, wpk, Cin, Cout, K);
    };
    auto bnst = [&](const float* X, int C, int lsh) {
        bn_stats<<<C, 256, 0, stream>>>(X, stats, C, lsh, 32);
    };
    auto bnap = [&](const float* X, const float* RES, float* Y,
                    const float* g, const float* b, int C, int lsh, int relu) {
        bn_apply<<<GB256, 256, 0, stream>>>(X, RES, Y, g, b, stats, C - 1, lsh, relu, TOT256);
    };

    // ======================= Encoder =======================
    conv1_direct<<<GB256, 256, 0, stream>>>(x, e_w1, e_b1, P128);
    bnst(P128, 128, 12);
    bnap(P128, nullptr, P128, e_bn1g, e_bn1b, 128, 12, 1);

    pack(e_w2, 128, 256, 4);
    conv1d_wmma<4, 2, 0><<<dim3(16, 16, 32), 256, 0, stream>>>(
        P128, wpk, e_b2, A, 128, 4096, 256, 2048, -1, 0, 1, 2, 1, 0);
    bnst(A, 256, 11);
    bnap(A, nullptr, A, e_bn2g, e_bn2b, 256, 11, 1);

    pack(e_w3, 256, 256, 3);
    conv1d_wmma<3, 1, 0><<<dim3(16, 16, 32), 256, 0, stream>>>(
        A, wpk, e_b3, Bf, 256, 2048, 256, 2048, -1, 0, 1, 0, 1, 0);
    bnst(Bf, 256, 11);
    bnap(Bf, nullptr, Bf, e_bn3g, e_bn3b, 256, 11, 0);     // h0 = Bf

    // res1 (input ReLU fused into conv)
    pack(er1_w1, 256, 256, 3);
    conv1d_wmma<3, 1, 1><<<dim3(16, 16, 32), 256, 0, stream>>>(
        Bf, wpk, nullptr, Cf, 256, 2048, 256, 2048, -1, 0, 1, 0, 1, 0);
    bnst(Cf, 256, 11);
    bnap(Cf, nullptr, Cf, er1_g1, er1_b1, 256, 11, 1);
    pack(er1_w2, 256, 256, 1);
    conv1d_wmma<1, 1, 0><<<dim3(16, 16, 32), 256, 0, stream>>>(
        Cf, wpk, nullptr, A, 256, 2048, 256, 2048, 0, 0, 0, 0, 1, 0);
    bnst(A, 256, 11);
    bnap(A, Bf, Bf, er1_g2, er1_b2, 256, 11, 0);

    // res2
    pack(er2_w1, 256, 256, 3);
    conv1d_wmma<3, 1, 1><<<dim3(16, 16, 32), 256, 0, stream>>>(
        Bf, wpk, nullptr, Cf, 256, 2048, 256, 2048, -1, 0, 1, 0, 1, 0);
    bnst(Cf, 256, 11);
    bnap(Cf, nullptr, Cf, er2_g1, er2_b1, 256, 11, 1);
    pack(er2_w2, 256, 256, 1);
    conv1d_wmma<1, 1, 0><<<dim3(16, 16, 32), 256, 0, stream>>>(
        Cf, wpk, nullptr, A, 256, 2048, 256, 2048, 0, 0, 0, 0, 1, 0);
    bnst(A, 256, 11);
    bnap(A, Bf, Bf, er2_g2, er2_b2, 256, 11, 0);

    // conv4 -> z_e
    pack(e_w4, 256, 64, 1);
    conv1d_wmma<1, 1, 0><<<dim3(16, 4, 32), 256, 0, stream>>>(
        Bf, wpk, e_b4, Z1, 256, 2048, 64, 2048, 0, 0, 0, 0, 1, 0);

    // ======================= Vector quantizer =======================
    enorm_k<<<2, 256, 0, stream>>>(Ecb, enormB);
    hipMemsetAsync(lossB, 0, 4, stream);
    vq_argmin<<<512, 256, 0, stream>>>(Z1, Ecb, enormB, idxI, out + 262145);
    vq_gather<<<16384, 256, 0, stream>>>(Z1, Ecb, idxI, Z2, lossB);
    finalize_loss<<<1, 1, 0, stream>>>(lossB, out + 262144);

    // ======================= Decoder =======================
    pack(d_w1, 64, 256, 3);
    conv1d_wmma<3, 1, 0><<<dim3(16, 16, 32), 256, 0, stream>>>(
        Z2, wpk, d_b1, A, 64, 2048, 256, 2048, -1, 0, 1, 0, 1, 0);
    bnst(A, 256, 11);
    bnap(A, nullptr, A, d_bn1g, d_bn1b, 256, 11, 0);       // h0 = A

    // dec res1
    pack(dr1_w1, 256, 256, 3);
    conv1d_wmma<3, 1, 1><<<dim3(16, 16, 32), 256, 0, stream>>>(
        A, wpk, nullptr, Cf, 256, 2048, 256, 2048, -1, 0, 1, 0, 1, 0);
    bnst(Cf, 256, 11);
    bnap(Cf, nullptr, Cf, dr1_g1, dr1_b1, 256, 11, 1);
    pack(dr1_w2, 256, 256, 1);
    conv1d_wmma<1, 1, 0><<<dim3(16, 16, 32), 256, 0, stream>>>(
        Cf, wpk, nullptr, Bf, 256, 2048, 256, 2048, 0, 0, 0, 0, 1, 0);
    bnst(Bf, 256, 11);
    bnap(Bf, A, A, dr1_g2, dr1_b2, 256, 11, 0);

    // dec res2
    pack(dr2_w1, 256, 256, 3);
    conv1d_wmma<3, 1, 1><<<dim3(16, 16, 32), 256, 0, stream>>>(
        A, wpk, nullptr, Cf, 256, 2048, 256, 2048, -1, 0, 1, 0, 1, 0);
    bnst(Cf, 256, 11);
    bnap(Cf, nullptr, Cf, dr2_g1, dr2_b1, 256, 11, 1);
    pack(dr2_w2, 256, 256, 1);
    conv1d_wmma<1, 1, 0><<<dim3(16, 16, 32), 256, 0, stream>>>(
        Cf, wpk, nullptr, Bf, 256, 2048, 256, 2048, 0, 0, 0, 0, 1, 0);
    bnst(Bf, 256, 11);
    bnap(Bf, A, A, dr2_g2, dr2_b2, 256, 11, 0);

    // convT1 as two stride-1 tap-major phase convs (R = 2*256 = 512)
    packT_weights<<<(256 * 128) / 256, 256, 0, stream>>>(d_wt1, wtp1, 256, 128);
    conv1d_wmma<2, 1, 0><<<dim3(16, 8, 32), 256, 0, stream>>>(
        A, wtp1,             d_bt1, P128, 256, 2048, 128, 4096, 0, -1, 0, 0, 2, 0);
    conv1d_wmma<2, 1, 0><<<dim3(16, 8, 32), 256, 0, stream>>>(
        A, wtp1 + 128 * 512, d_bt1, P128, 256, 2048, 128, 4096, 1,  0, 0, 0, 2, 1);
    bnst(P128, 128, 12);
    bnap(P128, nullptr, P128, d_bn2g, d_bn2b, 128, 12, 1);

    // convT2 -> x_recon
    convT2_direct<<<(32 * 8192) / 256, 256, 0, stream>>>(P128, d_wt2, d_bt2, out);
}
